// MultiHeadedAttention_23802708754966
// MI455X (gfx1250) — matrix operand order repaired and checked by reading, not compiled, not run on hardware
//
#include <hip/hip_runtime.h>
#include <stdint.h>

typedef __attribute__((ext_vector_type(16))) __bf16        v16bf;
typedef __attribute__((ext_vector_type(8)))  float         v8f;
typedef __attribute__((ext_vector_type(4)))  unsigned int  v4u;
typedef unsigned short ushort_t;

#define B_   4
#define S_   2048
#define D_   1024
#define H_   16
#define DK_  64
#define M_   (B_ * S_)   // 8192 rows for projection GEMMs
#define K_   D_          // 1024
#define N_   D_          // 1024

// bf16 fragment: 16 bf16 per lane = one A (16x32) or B (32x16) WMMA operand
union Frag {
    v16bf    v;
    ushort_t u[16];
    v4u      q[2];
};

__device__ __forceinline__ ushort_t f2bf(float x) {
    union { float f; unsigned int i; } c; c.f = x;
    unsigned int r = (c.i + 0x7FFFu + ((c.i >> 16) & 1u)) >> 16;  // RNE
    return (ushort_t)r;
}

// ---------------------------------------------------------------------------
// fp32 -> bf16 demotion, 8 elements / thread (b128 in, b128 out)
// n must be a multiple of 8 (true for all tensors here)
// ---------------------------------------------------------------------------
__global__ void f32_to_bf16_kernel(const float* __restrict__ src,
                                   ushort_t* __restrict__ dst, int n8) {
    int i = blockIdx.x * blockDim.x + threadIdx.x;
    if (i >= n8) return;
    const float* s = src + (size_t)i * 8;
    ushort_t o[8];
    #pragma unroll
    for (int j = 0; j < 8; ++j) o[j] = f2bf(s[j]);
    *(v4u*)(dst + (size_t)i * 8) = *(const v4u*)o;
}

// ---------------------------------------------------------------------------
// WMMA GEMM:  C[M_,N_] = A[M_,K_] (bf16) * W[N_,K_]^T (bf16) + bias
// One wave computes a 32x32 tile (2x2 of 16x16): 4 wmma per k-step on
// 2 A-frags + 2 B-frags -> 2 b128 loads per wmma, each frag used twice.
// MODE 0: store bf16 into [B,H,S,DK]   (Q, K projections)
// MODE 1: store bf16 into [B,H,DK,S]   (V projection, transposed for P*V)
// MODE 2: store fp32 into [M_,N_]      (output projection -> d_out)
// ---------------------------------------------------------------------------
template <int MODE>
__global__ __launch_bounds__(256) void gemm_wmma_kernel(
    const ushort_t* __restrict__ A,     // [M_, K_] bf16
    const ushort_t* __restrict__ W,     // [N_, K_] bf16 row-major (B = W^T)
    const float*    __restrict__ bias,  // [N_]
    void* __restrict__ outp)
{
    const int lane  = threadIdx.x & 31;
    const int wave  = threadIdx.x >> 5;
    const int half  = lane >> 4;       // 0: lanes 0-15, 1: lanes 16-31
    const int l15   = lane & 15;
    const int mbase = (blockIdx.y * 8 + wave) * 32;
    const int nbase = blockIdx.x * 32;

    v8f c[2][2] = {};
    #pragma unroll 2
    for (int k0 = 0; k0 < K_; k0 += 32) {
        Frag a[2], b[2];
        #pragma unroll
        for (int i = 0; i < 2; ++i) {
            // A 16x32 layout: lanes<16 -> K {0..7,16..23}; lanes>=16 -> {8..15,24..31}
            const ushort_t* ap = A + (size_t)(mbase + i * 16 + l15) * K_ + k0 + half * 8;
            a[i].q[0] = *(const v4u*)(ap);
            a[i].q[1] = *(const v4u*)(ap + 16);
            if (k0 + 32 < K_) __builtin_prefetch(ap + 32, 0, 0);  // global_prefetch_b8
        }
        #pragma unroll
        for (int j = 0; j < 2; ++j) {
            // B 32x16: lane = column n; lanes<16 hold K {0..7,16..23}, lanes>=16 hold {8..15,24..31}
            const ushort_t* wp = W + (size_t)(nbase + j * 16 + l15) * K_ + k0 + half * 8;
            b[j].q[0] = *(const v4u*)(wp);
            b[j].q[1] = *(const v4u*)(wp + 16);
            if (k0 + 32 < K_) __builtin_prefetch(wp + 32, 0, 0);
        }
        #pragma unroll
        for (int i = 0; i < 2; ++i)
            #pragma unroll
            for (int j = 0; j < 2; ++j)
                c[i][j] = __builtin_amdgcn_wmma_f32_16x16x32_bf16(
                              false, a[i].v, false, b[j].v, (short)0, c[i][j], false, false);
    }

    // C layout: VGPR r -> row M=r (lanes 0-15) / M=8+r (lanes 16-31); lane = N
    #pragma unroll
    for (int i = 0; i < 2; ++i) {
        #pragma unroll
        for (int j = 0; j < 2; ++j) {
            #pragma unroll
            for (int r = 0; r < 8; ++r) {
                const int m = mbase + i * 16 + r + half * 8;
                const int n = nbase + j * 16 + l15;
                float val = c[i][j][r] + bias[n];
                if (MODE == 2) {
                    ((float*)outp)[(size_t)m * N_ + n] = val;
                } else {
                    const int bb = m >> 11;          // m / S_
                    const int s  = m & (S_ - 1);
                    const int h  = n >> 6;           // n / DK_
                    const int dk = n & (DK_ - 1);
                    const ushort_t bv16 = f2bf(val);
                    if (MODE == 0)
                        ((ushort_t*)outp)[(((size_t)(bb * H_ + h)) * S_ + s) * DK_ + dk] = bv16;
                    else
                        ((ushort_t*)outp)[(((size_t)(bb * H_ + h)) * DK_ + dk) * S_ + s] = bv16;
                }
            }
        }
    }
}

// ---------------------------------------------------------------------------
// Flash attention: one wave owns a 32-row Q tile (two 16-row halves, 4 Q
// A-frags resident), streams 32-key tiles. Per tile: 8 score wmma + 8 P*V
// wmma on 16 b128 loads (K frags reused across both Q halves, V frags
// reused across both P halves). Online softmax in fp32.
// ---------------------------------------------------------------------------
#define LDSTRIDE 48   // ushorts per LDS row (96B -> keeps 16B alignment)

__global__ __launch_bounds__(256) void flash_attn_kernel(
    const ushort_t* __restrict__ Qp,   // [B,H,S,DK] bf16
    const ushort_t* __restrict__ Kp,   // [B,H,S,DK] bf16
    const ushort_t* __restrict__ Vt,   // [B,H,DK,S] bf16 (transposed)
    const int*      __restrict__ mask, // [B,S]
    ushort_t*       __restrict__ ctx)  // [B,S,D] bf16
{
    __shared__ __align__(16) ushort_t lds_p[8][32 * LDSTRIDE];

    const int lane = threadIdx.x & 31;
    const int wave = threadIdx.x >> 5;
    const int half = lane >> 4;
    const int l15  = lane & 15;

    const int bh    = blockIdx.y;
    const int b     = bh >> 4;          // / H_
    const int h     = bh & (H_ - 1);
    const int qbase = (blockIdx.x * 8 + wave) * 32;

    const ushort_t* Qb = Qp + (size_t)bh * S_ * DK_;
    const ushort_t* Kb = Kp + (size_t)bh * S_ * DK_;
    const ushort_t* Vb = Vt + (size_t)bh * DK_ * S_;

    // Q tile 32x64 = 2 q-halves x two 16x32 A-frags (resident for whole kernel)
    Frag qa[2][2];
    #pragma unroll
    for (int qh = 0; qh < 2; ++qh)
        #pragma unroll
        for (int cc = 0; cc < 2; ++cc) {
            const ushort_t* qp = Qb + (size_t)(qbase + qh * 16 + l15) * DK_ + cc * 32 + half * 8;
            qa[qh][cc].q[0] = *(const v4u*)(qp);
            qa[qh][cc].q[1] = *(const v4u*)(qp + 16);
        }

    v8f acc[2][4] = {};
    float m_r[2][8], l_r[2][8];
    #pragma unroll
    for (int qh = 0; qh < 2; ++qh)
        #pragma unroll
        for (int r = 0; r < 8; ++r) { m_r[qh][r] = -3.0e38f; l_r[qh][r] = 0.0f; }

    for (int kt = 0; kt < S_ / 32; ++kt) {
        // ---- K fragments: loaded once, reused by both Q halves ----
        Frag kb[2][2];           // [key group g][dk chunk cc]
        float maskadd[2];
        #pragma unroll
        for (int g = 0; g < 2; ++g) {
            const int key = kt * 32 + g * 16 + l15;   // this lane's key column
            maskadd[g] = (mask[b * S_ + key] == 0) ? -1.0e9f : 0.0f;
            #pragma unroll
            for (int cc = 0; cc < 2; ++cc) {
                const ushort_t* kp = Kb + (size_t)key * DK_ + cc * 32 + half * 8;
                kb[g][cc].q[0] = *(const v4u*)(kp);
                kb[g][cc].q[1] = *(const v4u*)(kp + 16);
            }
        }

        // ---- scores + online softmax per Q half; stage P into LDS ----
        #pragma unroll
        for (int qh = 0; qh < 2; ++qh) {
            v8f sg[2];
            #pragma unroll
            for (int g = 0; g < 2; ++g) {
                v8f s = {};
                s = __builtin_amdgcn_wmma_f32_16x16x32_bf16(
                        false, qa[qh][0].v, false, kb[g][0].v, (short)0, s, false, false);
                s = __builtin_amdgcn_wmma_f32_16x16x32_bf16(
                        false, qa[qh][1].v, false, kb[g][1].v, (short)0, s, false, false);
                sg[g] = s * 0.125f + maskadd[g];       // 1/sqrt(DK), then mask
            }
            #pragma unroll
            for (int r = 0; r < 8; ++r) {
                float t = fmaxf(sg[0][r], sg[1][r]);
                t = fmaxf(t, __shfl_xor(t, 1, 32));
                t = fmaxf(t, __shfl_xor(t, 2, 32));
                t = fmaxf(t, __shfl_xor(t, 4, 32));
                t = fmaxf(t, __shfl_xor(t, 8, 32));
                const float nm    = fmaxf(m_r[qh][r], t);
                const float alpha = __expf(m_r[qh][r] - nm);
                const float p0 = __expf(sg[0][r] - nm);
                const float p1 = __expf(sg[1][r] - nm);
                float rs = p0 + p1;
                rs += __shfl_xor(rs, 1, 32);
                rs += __shfl_xor(rs, 2, 32);
                rs += __shfl_xor(rs, 4, 32);
                rs += __shfl_xor(rs, 8, 32);
                l_r[qh][r] = l_r[qh][r] * alpha + rs;
                m_r[qh][r] = nm;
                #pragma unroll
                for (int t4 = 0; t4 < 4; ++t4) acc[qh][t4][r] *= alpha;
                const int row = qh * 16 + r + half * 8;
                lds_p[wave][row * LDSTRIDE + l15]      = f2bf(p0);
                lds_p[wave][row * LDSTRIDE + 16 + l15] = f2bf(p1);
            }
        }

        // ---- P fragments (transposed reload from LDS), one per Q half ----
        Frag pa[2];
        #pragma unroll
        for (int qh = 0; qh < 2; ++qh) {
            const ushort_t* lp = &lds_p[wave][(qh * 16 + l15) * LDSTRIDE + half * 8];
            pa[qh].q[0] = *(const v4u*)(lp);
            pa[qh].q[1] = *(const v4u*)(lp + 16);
        }

        // ---- O += P * V: each V frag loaded once, used by both P halves ----
        #pragma unroll
        for (int t4 = 0; t4 < 4; ++t4) {
            Frag vb;  // B-frag: column n = dk = t4*16+l15; rows = 32 keys (table order)
            const ushort_t* vp = Vb + (size_t)(t4 * 16 + l15) * S_ + kt * 32 + half * 8;
            vb.q[0] = *(const v4u*)(vp);
            vb.q[1] = *(const v4u*)(vp + 16);
            #pragma unroll
            for (int qh = 0; qh < 2; ++qh)
                acc[qh][t4] = __builtin_amdgcn_wmma_f32_16x16x32_bf16(
                                  false, pa[qh].v, false, vb.v, (short)0, acc[qh][t4], false, false);
        }
    }

    // ---- normalize + store ctx [B,S,D] bf16 ----
    #pragma unroll
    for (int qh = 0; qh < 2; ++qh)
        #pragma unroll
        for (int t4 = 0; t4 < 4; ++t4)
            #pragma unroll
            for (int r = 0; r < 8; ++r) {
                const float val = acc[qh][t4][r] / l_r[qh][r];
                const int s = qbase + qh * 16 + r + half * 8;
                const int d = h * DK_ + t4 * 16 + l15;
                ctx[((size_t)(b * S_ + s)) * D_ + d] = f2bf(val);
            }
}

// ---------------------------------------------------------------------------
// Host-side orchestration
// ---------------------------------------------------------------------------
extern "C" void kernel_launch(void* const* d_in, const int* in_sizes, int n_in,
                              void* d_out, int out_size, void* d_ws, size_t ws_size,
                              hipStream_t stream) {
    (void)in_sizes; (void)n_in; (void)out_size; (void)ws_size;

    const float* query = (const float*)d_in[0];
    const float* key_i = (const float*)d_in[1];
    const float* value = (const float*)d_in[2];
    const int*   mask  = (const int*)  d_in[3];
    const float* Wq = (const float*)d_in[4];
    const float* bq = (const float*)d_in[5];
    const float* Wk = (const float*)d_in[6];
    const float* bk = (const float*)d_in[7];
    const float* Wv = (const float*)d_in[8];
    const float* bv = (const float*)d_in[9];
    const float* Wo = (const float*)d_in[10];
    const float* bo = (const float*)d_in[11];

    char*  ws  = (char*)d_ws;
    size_t off = 0;
    auto alloc = [&](size_t bytes) -> void* {
        void* p = ws + off;
        off += (bytes + 255) & ~(size_t)255;
        return p;
    };

    const size_t MD = (size_t)M_ * K_;   // 8192*1024 activation elements
    const size_t WD = (size_t)N_ * K_;   // 1024*1024 weight elements

    ushort_t* actQ = (ushort_t*)alloc(MD * 2);
    ushort_t* actK = (ushort_t*)alloc(MD * 2);
    ushort_t* actV = (ushort_t*)alloc(MD * 2);
    ushort_t* wqb  = (ushort_t*)alloc(WD * 2);
    ushort_t* wkb  = (ushort_t*)alloc(WD * 2);
    ushort_t* wvb  = (ushort_t*)alloc(WD * 2);
    ushort_t* wob  = (ushort_t*)alloc(WD * 2);
    ushort_t* Qp   = (ushort_t*)alloc(MD * 2);   // [B,H,S,DK]
    ushort_t* Kp   = (ushort_t*)alloc(MD * 2);   // [B,H,S,DK]
    ushort_t* Vt   = (ushort_t*)alloc(MD * 2);   // [B,H,DK,S]
    ushort_t* ctx  = (ushort_t*)alloc(MD * 2);   // [B,S,D]

    const int cb  = 256;
    const int nM8 = (int)(MD / 8), nW8 = (int)(WD / 8);
    f32_to_bf16_kernel<<<(nM8 + cb - 1) / cb, cb, 0, stream>>>(query, actQ, nM8);
    f32_to_bf16_kernel<<<(nM8 + cb - 1) / cb, cb, 0, stream>>>(key_i, actK, nM8);
    f32_to_bf16_kernel<<<(nM8 + cb - 1) / cb, cb, 0, stream>>>(value, actV, nM8);
    f32_to_bf16_kernel<<<(nW8 + cb - 1) / cb, cb, 0, stream>>>(Wq, wqb, nW8);
    f32_to_bf16_kernel<<<(nW8 + cb - 1) / cb, cb, 0, stream>>>(Wk, wkb, nW8);
    f32_to_bf16_kernel<<<(nW8 + cb - 1) / cb, cb, 0, stream>>>(Wv, wvb, nW8);
    f32_to_bf16_kernel<<<(nW8 + cb - 1) / cb, cb, 0, stream>>>(Wo, wob, nW8);

    const dim3 gg(N_ / 32, M_ / 32 / 8);   // (32, 32), 256 threads = 8 waves
    gemm_wmma_kernel<0><<<gg, 256, 0, stream>>>(actQ, wqb, bq, Qp);
    gemm_wmma_kernel<0><<<gg, 256, 0, stream>>>(actK, wkb, bk, Kp);
    gemm_wmma_kernel<1><<<gg, 256, 0, stream>>>(actV, wvb, bv, Vt);

    const dim3 ga(S_ / 32 / 8, B_ * H_);   // (8, 64)
    flash_attn_kernel<<<ga, 256, 0, stream>>>(Qp, Kp, Vt, mask, ctx);

    gemm_wmma_kernel<2><<<gg, 256, 0, stream>>>(ctx, wob, bo, (float*)d_out);
}
